// HEAD_47459388620826
// MI455X (gfx1250) — compile-verified, matrix operand order checked by reading, not run on hardware
//
#include <hip/hip_runtime.h>
#include <hip/hip_bf16.h>

typedef __attribute__((ext_vector_type(16))) __bf16 v16bf;
typedef __attribute__((ext_vector_type(8)))  float  v8f;

#define ENC_NEG_INF 0x007FFFFFu

__device__ __forceinline__ unsigned encf(float f) {
    unsigned u = __float_as_uint(f);
    return (u >> 31) ? ~u : (u | 0x80000000u);
}
__device__ __forceinline__ float decf(unsigned e) {
    return (e >> 31) ? __uint_as_float(e & 0x7FFFFFFFu) : __uint_as_float(~e);
}

// ---------------------------------------------------------------------------
// Pack [W1;W2;zeros] (rows of 66) and [b1;b2;zeros] into Dpad-row buffers.
// ---------------------------------------------------------------------------
__global__ void pack2_kernel(const float* __restrict__ W1, const float* __restrict__ b1, int D1,
                             const float* __restrict__ W2, const float* __restrict__ b2, int D2,
                             float* __restrict__ Wd, float* __restrict__ bd, int Dpad)
{
    int idx = blockIdx.x * 256 + threadIdx.x;
    int total = Dpad * 66;
    if (idx < total) {
        int d = idx / 66, k = idx - d * 66;
        float v = 0.f;
        if (d < D1) v = W1[d * 66 + k];
        else if (d < D1 + D2) v = W2[(d - D1) * 66 + k];
        Wd[idx] = v;
    } else if (idx < total + Dpad) {
        int d = idx - total;
        float v = 0.f;
        if (d < D1) v = b1[d];
        else if (d < D1 + D2) v = b2[d - D1];
        bd[d] = v;
    }
}

// ---------------------------------------------------------------------------
// Per-node GEMM via WMMA (bf16x3 split for ~fp32 accuracy):
//   A[i][d] = F[i][:64] @ W[d][:64] + pos2[i][0]*W[d][64] + pos2[i][1]*W[d][65]
// Also initializes M to enc(-inf).
// Block: 256 thr = 8 waves. 64 rows/block; wave (wv&3) owns a 16-row strip,
// phase (wv>>2) splits the Dpad/16 column tiles.
// ---------------------------------------------------------------------------
__global__ void gemm_node_kernel(const float* __restrict__ F, int Fstride,
                                 const float* __restrict__ pos,
                                 const float* __restrict__ W,   // Dpad x 66
                                 int Dpad,
                                 float* __restrict__ A, unsigned* __restrict__ M,
                                 int N)
{
    __shared__ float  Fls[64][64];
    __shared__ __bf16 Whi[128 * 64];
    __shared__ __bf16 Wlo[128 * 64];
    __shared__ float  PosT[64][2];
    __shared__ float  Wp[128][2];

    const int tid = threadIdx.x;
    const int rowBase = blockIdx.x * 64;

    // Stage F tile (64 rows x 64 cols f32), zero-padded past N.
    for (int i = tid; i < 64 * 16; i += 256) {
        int r = i >> 4;
        int c4 = (i & 15) * 4;
        int row = rowBase + r;
        float4 v = make_float4(0.f, 0.f, 0.f, 0.f);
        if (row < N) v = *(const float4*)(F + (size_t)row * Fstride + c4);
        *(float4*)&Fls[r][c4] = v;
    }
    for (int i = tid; i < 64; i += 256) {
        int row = rowBase + i;
        PosT[i][0] = (row < N) ? pos[row * 3 + 0] : 0.f;
        PosT[i][1] = (row < N) ? pos[row * 3 + 1] : 0.f;
    }
    // Stage W as bf16 hi/lo split.
    for (int i = tid; i < Dpad * 64; i += 256) {
        int d = i >> 6, k = i & 63;
        float w = W[d * 66 + k];
        __bf16 hi = (__bf16)w;
        Whi[i] = hi;
        Wlo[i] = (__bf16)(w - (float)hi);
    }
    for (int i = tid; i < Dpad; i += 256) {
        Wp[i][0] = W[i * 66 + 64];
        Wp[i][1] = W[i * 66 + 65];
    }
    __syncthreads();

    const int lane  = tid & 31;
    const int wv    = tid >> 5;
    const int strip = wv & 3;
    const int ctPhase = wv >> 2;
    const int mrow  = lane & 15;
    const int khalf = (lane >> 4) ? 8 : 0;

    const int kconst[16] = {0,1,2,3,4,5,6,7,16,17,18,19,20,21,22,23};

    // A fragments (16x32 bf16, ISA layout), hi+lo, for both K-tiles.
    v16bf ahi[2], alo[2];
#pragma unroll
    for (int kt = 0; kt < 2; ++kt) {
#pragma unroll
        for (int j = 0; j < 16; ++j) {
            int K = kt * 32 + kconst[j] + khalf;
            float v = Fls[strip * 16 + mrow][K];
            __bf16 h = (__bf16)v;
            ahi[kt][j] = h;
            alo[kt][j] = (__bf16)(v - (float)h);
        }
    }

    int nct = Dpad >> 4;
    for (int ct = ctPhase; ct < nct; ct += 2) {
        v16bf bhi[2], blo[2];
#pragma unroll
        for (int kt = 0; kt < 2; ++kt) {
#pragma unroll
            for (int j = 0; j < 16; ++j) {
                int K = kt * 32 + kconst[j] + khalf;
                int d = ct * 16 + mrow;
                bhi[kt][j] = Whi[d * 64 + K];
                blo[kt][j] = Wlo[d * 64 + K];
            }
        }
        v8f acc = {};
#pragma unroll
        for (int kt = 0; kt < 2; ++kt) {
            acc = __builtin_amdgcn_wmma_f32_16x16x32_bf16(false, ahi[kt], false, bhi[kt], (short)0, acc, false, false);
            acc = __builtin_amdgcn_wmma_f32_16x16x32_bf16(false, ahi[kt], false, blo[kt], (short)0, acc, false, false);
            acc = __builtin_amdgcn_wmma_f32_16x16x32_bf16(false, alo[kt], false, bhi[kt], (short)0, acc, false, false);
        }
        int col = ct * 16 + mrow;
        float wpx = Wp[col][0], wpy = Wp[col][1];
#pragma unroll
        for (int r = 0; r < 8; ++r) {
            int mloc = r + ((lane >> 4) ? 8 : 0);
            int row = rowBase + strip * 16 + mloc;
            if (row < N) {
                float p = PosT[strip * 16 + mloc][0] * wpx + PosT[strip * 16 + mloc][1] * wpy;
                size_t o = (size_t)row * Dpad + col;
                A[o] = acc[r] + p;
                M[o] = ENC_NEG_INF;
            }
        }
    }
}

// ---------------------------------------------------------------------------
// Edge scatter-max: M[dst,:] = max(M[dst,:], A[src,:]) on encoded uints.
// One thread per (edge, 4-channel quad).
// ---------------------------------------------------------------------------
__global__ void edge_max_kernel(const float* __restrict__ A, unsigned* __restrict__ M,
                                const int* __restrict__ src, const int* __restrict__ dst,
                                int E, int Dpad, int DS4)
{
    int idx = blockIdx.x * 256 + threadIdx.x;
    int total = E * DS4;
    if (idx >= total) return;
    int e = idx / DS4;
    int q = idx - e * DS4;
    int s = src[e], d = dst[e];
    const float4 v = *(const float4*)(A + (size_t)s * Dpad + q * 4);
    unsigned* mp = M + (size_t)d * Dpad + q * 4;
    atomicMax(mp + 0, encf(v.x));
    atomicMax(mp + 1, encf(v.y));
    atomicMax(mp + 2, encf(v.z));
    atomicMax(mp + 3, encf(v.w));
}

// ---------------------------------------------------------------------------
// BN stats: y = (M finite ? b - pos2@Wp + dec(M) : 0); sum & sumsq per channel.
// ---------------------------------------------------------------------------
__global__ void conv_stats_kernel(const unsigned* __restrict__ M, int Dpad, int chanOff,
                                  const float* __restrict__ W, const float* __restrict__ bvec,
                                  const float* __restrict__ pos,
                                  float* __restrict__ sums, int N)
{
    __shared__ float red[256];
    int tid = threadIdx.x;
    int c = tid & 63;
    int g = tid >> 6;
    int ci = chanOff + c;
    float wx = W[ci * 66 + 64], wy = W[ci * 66 + 65], bb = bvec[ci];
    float s = 0.f, sq = 0.f;
    for (int r = blockIdx.x * 4 + g; r < N; r += gridDim.x * 4) {
        unsigned mv = M[(size_t)r * Dpad + ci];
        float y = 0.f;
        if (mv != ENC_NEG_INF)
            y = bb - (pos[r * 3] * wx + pos[r * 3 + 1] * wy) + decf(mv);
        s += y; sq += y * y;
    }
    red[tid] = s;
    __syncthreads();
    if (g == 0) atomicAdd(&sums[c], red[c] + red[64 + c] + red[128 + c] + red[192 + c]);
    __syncthreads();
    red[tid] = sq;
    __syncthreads();
    if (g == 0) atomicAdd(&sums[64 + c], red[c] + red[64 + c] + red[128 + c] + red[192 + c]);
}

// ---------------------------------------------------------------------------
// BN apply + ReLU: out = relu((y-mean)*rsqrt(var+eps)*gamma + beta)
// ---------------------------------------------------------------------------
__global__ void conv_bn_apply_kernel(const unsigned* __restrict__ M, int Dpad, int chanOff,
                                     const float* __restrict__ W, const float* __restrict__ bvec,
                                     const float* __restrict__ pos,
                                     const float* __restrict__ sums,
                                     const float* __restrict__ gamma, const float* __restrict__ beta,
                                     float* __restrict__ out, int outStride, int outOff, int N)
{
    int idx = blockIdx.x * 256 + threadIdx.x;
    if (idx >= N * 64) return;
    int r = idx >> 6, c = idx & 63;
    int ci = chanOff + c;
    unsigned mv = M[(size_t)r * Dpad + ci];
    float y = 0.f;
    if (mv != ENC_NEG_INF)
        y = bvec[ci] - (pos[r * 3] * W[ci * 66 + 64] + pos[r * 3 + 1] * W[ci * 66 + 65]) + decf(mv);
    float invN = 1.f / (float)N;
    float mean = sums[c] * invN;
    float var  = sums[64 + c] * invN - mean * mean;
    float sc   = gamma[c] * rsqrtf(var + 1e-5f);
    out[(size_t)r * outStride + outOff + c] = fmaxf(0.f, (y - mean) * sc + beta[c]);
}

// ---------------------------------------------------------------------------
// Final conv output (no BN): out = finite ? b - pos2@Wp + dec(M) : 0
// ---------------------------------------------------------------------------
__global__ void conv_out_kernel(const unsigned* __restrict__ M, int Dpad, int chanOff, int Dused,
                                const float* __restrict__ W, const float* __restrict__ bvec,
                                const float* __restrict__ pos,
                                float* __restrict__ out, int outStride, int N)
{
    int idx = blockIdx.x * 256 + threadIdx.x;
    if (idx >= N * Dused) return;
    int r = idx / Dused, d = idx - r * Dused;
    int ci = chanOff + d;
    unsigned mv = M[(size_t)r * Dpad + ci];
    float y = 0.f;
    if (mv != ENC_NEG_INF)
        y = bvec[ci] - (pos[r * 3] * W[ci * 66 + 64] + pos[r * 3 + 1] * W[ci * 66 + 65]) + decf(mv);
    out[(size_t)r * outStride + d] = y;
}

// ---------------------------------------------------------------------------
extern "C" void kernel_launch(void* const* d_in, const int* in_sizes, int n_in,
                              void* d_out, int out_size, void* d_ws, size_t ws_size,
                              hipStream_t stream)
{
    const float* x       = (const float*)d_in[0];
    const float* pos     = (const float*)d_in[1];
    const int*   edges   = (const int*)d_in[2];   // JAX int64 w/o x64 => int32
    const float* W_stem  = (const float*)d_in[3];
    const float* b_stem  = (const float*)d_in[4];
    const float* W_c1    = (const float*)d_in[5];
    const float* b_c1    = (const float*)d_in[6];
    const float* W_c2    = (const float*)d_in[7];
    const float* b_c2    = (const float*)d_in[8];
    const float* W_regr  = (const float*)d_in[9];
    const float* b_regr  = (const float*)d_in[10];
    const float* W_cls   = (const float*)d_in[11];
    const float* b_cls   = (const float*)d_in[12];
    const float* W_obj   = (const float*)d_in[13];
    const float* b_obj   = (const float*)d_in[14];
    const float* g_stem  = (const float*)d_in[15];
    const float* be_stem = (const float*)d_in[16];
    const float* g_c1    = (const float*)d_in[17];
    const float* be_c1   = (const float*)d_in[18];
    const float* g_c2    = (const float*)d_in[19];
    const float* be_c2   = (const float*)d_in[20];

    const int N = in_sizes[0] / 64;
    const int E = in_sizes[2] / 2;
    const int* src = edges;
    const int* dst = edges + E;

    // Workspace arena
    char* ws = (char*)d_ws;
    size_t off = 0;
    auto alloc = [&](size_t bytes) { size_t r = off; off = (off + bytes + 255) & ~(size_t)255; return r; };
    float*    h     = (float*)(ws + alloc((size_t)N * 64 * 4));
    float*    x12   = (float*)(ws + alloc((size_t)N * 128 * 4));
    float*    bufA  = (float*)(ws + alloc((size_t)N * 128 * 4));
    unsigned* bufM  = (unsigned*)(ws + alloc((size_t)N * 128 * 4));
    float*    Wp12  = (float*)(ws + alloc(128 * 66 * 4));
    float*    bp12  = (float*)(ws + alloc(128 * 4));
    float*    Wpro  = (float*)(ws + alloc(16 * 66 * 4));
    float*    bpro  = (float*)(ws + alloc(16 * 4));
    float*    Wpcls = (float*)(ws + alloc(112 * 66 * 4));
    float*    bpcls = (float*)(ws + alloc(112 * 4));
    float*    sumsS = (float*)(ws + alloc(128 * 4));
    float*    sums1 = (float*)(ws + alloc(128 * 4));
    float*    sums2 = (float*)(ws + alloc(128 * 4));
    (void)ws_size;

    hipMemsetAsync(sumsS, 0, 3 * 128 * 4, stream);  // sumsS/sums1/sums2 contiguous

    // Pack weights
    pack2_kernel<<<(128 * 67 + 255) / 256, 256, 0, stream>>>(W_c1, b_c1, 64, W_c2, b_c2, 64, Wp12, bp12, 128);
    pack2_kernel<<<(16 * 67 + 255) / 256, 256, 0, stream>>>(W_regr, b_regr, 4, W_obj, b_obj, 1, Wpro, bpro, 16);
    pack2_kernel<<<(112 * 67 + 255) / 256, 256, 0, stream>>>(W_cls, b_cls, 101, nullptr, nullptr, 0, Wpcls, bpcls, 112);

    const int gemmBlocks  = (N + 63) / 64;
    const int statsBlocks = 1024;
    const int applyBlocks = (N * 64 + 255) / 256;

    // ---- STEM ----
    gemm_node_kernel<<<gemmBlocks, 256, 0, stream>>>(x, 64, pos, W_stem, 64, bufA, bufM, N);
    edge_max_kernel<<<(E * 16 + 255) / 256, 256, 0, stream>>>(bufA, bufM, src, dst, E, 64, 16);
    conv_stats_kernel<<<statsBlocks, 256, 0, stream>>>(bufM, 64, 0, W_stem, b_stem, pos, sumsS, N);
    conv_bn_apply_kernel<<<applyBlocks, 256, 0, stream>>>(bufM, 64, 0, W_stem, b_stem, pos, sumsS,
                                                          g_stem, be_stem, h, 64, 0, N);

    // ---- CONV1 + CONV2 (fused, D=128) ----
    gemm_node_kernel<<<gemmBlocks, 256, 0, stream>>>(h, 64, pos, Wp12, 128, bufA, bufM, N);
    edge_max_kernel<<<(E * 32 + 255) / 256, 256, 0, stream>>>(bufA, bufM, src, dst, E, 128, 32);
    conv_stats_kernel<<<statsBlocks, 256, 0, stream>>>(bufM, 128, 0, Wp12, bp12, pos, sums1, N);
    conv_stats_kernel<<<statsBlocks, 256, 0, stream>>>(bufM, 128, 64, Wp12, bp12, pos, sums2, N);
    conv_bn_apply_kernel<<<applyBlocks, 256, 0, stream>>>(bufM, 128, 0, Wp12, bp12, pos, sums1,
                                                          g_c1, be_c1, x12, 128, 0, N);
    conv_bn_apply_kernel<<<applyBlocks, 256, 0, stream>>>(bufM, 128, 64, Wp12, bp12, pos, sums2,
                                                          g_c2, be_c2, x12, 128, 64, N);

    // ---- REGR + OBJ (fused, D=16 padded, input x1 = x12[:, :64]) ----
    gemm_node_kernel<<<gemmBlocks, 256, 0, stream>>>(x12, 128, pos, Wpro, 16, bufA, bufM, N);
    edge_max_kernel<<<(E * 2 + 255) / 256, 256, 0, stream>>>(bufA, bufM, src, dst, E, 16, 2);
    float* out_cls = (float*)d_out;
    float* out_reg = out_cls + (size_t)N * 101;
    float* out_obj = out_reg + (size_t)N * 4;
    conv_out_kernel<<<(N * 4 + 255) / 256, 256, 0, stream>>>(bufM, 16, 0, 4, Wpro, bpro, pos, out_reg, 4, N);
    conv_out_kernel<<<(N + 255) / 256, 256, 0, stream>>>(bufM, 16, 4, 1, Wpro, bpro, pos, out_obj, 1, N);

    // ---- CLS (D=112 padded, input x2 = x12[:, 64:]) ----
    gemm_node_kernel<<<gemmBlocks, 256, 0, stream>>>(x12 + 64, 128, pos, Wpcls, 112, bufA, bufM, N);
    edge_max_kernel<<<(E * 26 + 255) / 256, 256, 0, stream>>>(bufA, bufM, src, dst, E, 112, 26);
    conv_out_kernel<<<(N * 101 + 255) / 256, 256, 0, stream>>>(bufM, 112, 0, 101, Wpcls, bpcls, pos, out_cls, 101, N);
}
